// RAN_87016037417231
// MI455X (gfx1250) — matrix operand order repaired and checked by reading, not compiled, not run on hardware
//
#include <hip/hip_runtime.h>
#include <hip/hip_bf16.h>
#include <cstdint>

// Problem sizes (fixed by the reference)
#define S_LEN 128
#define B_DIM 32
#define H_DIM 1024
#define E_DIM 1024
#define V_DIM 32000
#define BH    (B_DIM * H_DIM)            // 32768
#define MROWS (S_LEN * B_DIM)            // 4096
#define HH    (H_DIM * H_DIM)            // 1048576
#define LOGN  ((size_t)MROWS * V_DIM)    // 131072000

typedef __attribute__((ext_vector_type(16))) __bf16 bf16x16;
typedef __attribute__((ext_vector_type(8)))  __bf16 bf16x8;
typedef __attribute__((ext_vector_type(4)))  __bf16 bf16x4;
typedef __attribute__((ext_vector_type(8)))  float  f32x8;
typedef __attribute__((ext_vector_type(4)))  unsigned int u32x4;
typedef __attribute__((ext_vector_type(4)))  int    i32x4;
typedef __attribute__((ext_vector_type(8)))  int    i32x8;

#if defined(__has_builtin)
#if __has_builtin(__builtin_amdgcn_tensor_load_to_lds) && \
    __has_builtin(__builtin_amdgcn_s_wait_tensorcnt)
#define HAVE_TDM 1
#endif
#endif

// ---------------------------------------------------------------------------
// WMMA helpers (CDNA5: V_WMMA_F32_16X16X32_BF16, wave32)
// ---------------------------------------------------------------------------
__device__ __forceinline__ f32x8 wmma_bf16(bf16x16 a, bf16x16 b, f32x8 c) {
  // 8 args: (neg_a, A, neg_b, B, c_mod, C, reuse_a, reuse_b)
  return __builtin_amdgcn_wmma_f32_16x16x32_bf16(false, a, false, b, (short)0, c,
                                                 false, false);
}

// A fragment: 16x32 bf16 tile, row-major source with stride lda.
// ISA layout: lane L -> row = L%16; half=L/16 selects K interleave:
//   half=0: K in [0..7] and [16..23]; half=1: K in [8..15] and [24..31].
__device__ __forceinline__ bf16x16
load_a_frag(const __bf16* __restrict__ A, int lda, int row0, int k0, int lane) {
  const int half = lane >> 4;
  const int r    = lane & 15;
  const __bf16* p = A + (size_t)(row0 + r) * lda + (k0 + half * 8);
  bf16x8 c1 = *reinterpret_cast<const bf16x8*>(p);
  bf16x8 c2 = *reinterpret_cast<const bf16x8*>(p + 16);
  return __builtin_shufflevector(c1, c2, 0, 1, 2, 3, 4, 5, 6, 7,
                                 8, 9, 10, 11, 12, 13, 14, 15);
}

// B fragment: 32x16 (KxN) built from W[N][K] row-major (i.e. B^T in memory).
__device__ __forceinline__ bf16x16
load_b_frag(const __bf16* __restrict__ W, int ldb, int col0, int k0, int lane) {
  const int half = lane >> 4;
  const int c    = lane & 15;
  const __bf16* p = W + (size_t)(col0 + c) * ldb + (k0 + half * 8);
  bf16x8 c1 = *reinterpret_cast<const bf16x8*>(p);
  bf16x8 c2 = *reinterpret_cast<const bf16x8*>(p + 16);
  return __builtin_shufflevector(c1, c2, 0, 1, 2, 3, 4, 5, 6, 7,
                                 8, 9, 10, 11, 12, 13, 14, 15);
}

// Same but converting f32 weights (h2o_w stays f32 in L2; cvt in registers).
__device__ __forceinline__ bf16x16
load_b_frag_f32(const float* __restrict__ W, int ldb, int col0, int k0, int lane) {
  const int half = lane >> 4;
  const int c    = lane & 15;
  const float* p = W + (size_t)(col0 + c) * ldb + (k0 + half * 8);
  const float4* q = reinterpret_cast<const float4*>(p);
  float4 v0 = q[0], v1 = q[1], v2 = q[4], v3 = q[5];
  bf16x16 o = {(__bf16)v0.x, (__bf16)v0.y, (__bf16)v0.z, (__bf16)v0.w,
               (__bf16)v1.x, (__bf16)v1.y, (__bf16)v1.z, (__bf16)v1.w,
               (__bf16)v2.x, (__bf16)v2.y, (__bf16)v2.z, (__bf16)v2.w,
               (__bf16)v3.x, (__bf16)v3.y, (__bf16)v3.z, (__bf16)v3.w};
  return o;
}

// ---------------------------------------------------------------------------
// Weight f32 -> bf16 conversion (8 x 1024x1024 matrices, done once per launch)
// ---------------------------------------------------------------------------
__global__ __launch_bounds__(256) void cvt_weights_kernel(
    const float* __restrict__ w0, const float* __restrict__ w1,
    const float* __restrict__ w2, const float* __restrict__ w3,
    const float* __restrict__ w4, const float* __restrict__ w5,
    const float* __restrict__ w6, const float* __restrict__ w7,
    __bf16* __restrict__ dst) {
  const float* tab[8] = {w0, w1, w2, w3, w4, w5, w6, w7};
  size_t idx = ((size_t)blockIdx.x * 256 + threadIdx.x) * 4;  // over 8*HH elems
  int which = (int)(idx >> 20);
  size_t off = idx & ((1u << 20) - 1);
  float4 v = *reinterpret_cast<const float4*>(tab[which] + off);
  bf16x4 o = {(__bf16)v.x, (__bf16)v.y, (__bf16)v.z, (__bf16)v.w};
  *reinterpret_cast<bf16x4*>(dst + idx) = o;
}

// Embedding gather + bf16 convert: x1[s,b,:] = bf16(emb[word[s,b],:])
__global__ __launch_bounds__(256) void gather_embed_kernel(
    const int* __restrict__ word, const float* __restrict__ emb,
    __bf16* __restrict__ x1) {
  int sb = blockIdx.x;                       // s*B + b, 0..4095
  int row = word[sb];
  int e = threadIdx.x * 4;                   // 256*4 = 1024 = E
  float4 v = *reinterpret_cast<const float4*>(emb + (size_t)row * E_DIM + e);
  bf16x4 o = {(__bf16)v.x, (__bf16)v.y, (__bf16)v.z, (__bf16)v.w};
  *reinterpret_cast<bf16x4*>(x1 + (size_t)sb * E_DIM + e) = o;
}

__global__ __launch_bounds__(256) void init_latent_kernel(
    const float* __restrict__ lat_in, float* __restrict__ lat_f32,
    __bf16* __restrict__ lat_bf16) {
  int i = blockIdx.x * 256 + threadIdx.x;    // 128 blocks -> 32768
  float v = lat_in[i];
  lat_f32[i] = v;
  lat_bf16[i] = (__bf16)v;
}

// ---------------------------------------------------------------------------
// GEMM: out[M,N] = A[M,K](bf16) * W[N,K]^T(bf16) + bias
// Block = 256 threads = 8 waves; block tile 128x128; wave tile 32x64.
// ---------------------------------------------------------------------------
__global__ __launch_bounds__(256) void gemm_bf16_kernel(
    const __bf16* __restrict__ A, const __bf16* __restrict__ W,
    const float* __restrict__ bias, float* __restrict__ out,
    int M, int N, int K) {
  const int lane = threadIdx.x & 31;
  const int wave = threadIdx.x >> 5;   // 0..7
  const int wm = wave & 3;             // 4 waves along M
  const int wn = wave >> 2;            // 2 waves along N
  const int row0 = blockIdx.x * 128 + wm * 32;
  const int col0 = blockIdx.y * 128 + wn * 64;

  f32x8 c[2][4] = {};
  for (int k0 = 0; k0 < K; k0 += 32) {
    __builtin_prefetch(A + (size_t)(row0 + (lane & 15)) * K + k0 + 64, 0, 3);
    bf16x16 a0 = load_a_frag(A, K, row0,      k0, lane);
    bf16x16 a1 = load_a_frag(A, K, row0 + 16, k0, lane);
#pragma unroll
    for (int j = 0; j < 4; ++j) {
      bf16x16 b = load_b_frag(W, K, col0 + j * 16, k0, lane);
      c[0][j] = wmma_bf16(a0, b, c[0][j]);
      c[1][j] = wmma_bf16(a1, b, c[1][j]);
    }
  }

  const int half = lane >> 4, nn = lane & 15;
#pragma unroll
  for (int i = 0; i < 2; ++i)
#pragma unroll
    for (int j = 0; j < 4; ++j) {
      int col = col0 + j * 16 + nn;
      float bb = bias ? bias[col] : 0.0f;
#pragma unroll
      for (int r = 0; r < 8; ++r) {
        int row = row0 + i * 16 + 8 * half + r;
        out[(size_t)row * N + col] = c[i][j][r] + bb;
      }
    }
}

// Logits GEMM: A=l2_out(bf16), W=h2o_w(f32, cvt in regs), N=32000.
__global__ __launch_bounds__(256) void gemm_logits_kernel(
    const __bf16* __restrict__ A, const float* __restrict__ W,
    const float* __restrict__ bias, float* __restrict__ out,
    int M, int N, int K) {
  const int lane = threadIdx.x & 31;
  const int wave = threadIdx.x >> 5;
  const int wm = wave & 3;
  const int wn = wave >> 2;
  const int row0 = blockIdx.x * 128 + wm * 32;
  const int col0 = blockIdx.y * 128 + wn * 64;

  f32x8 c[2][4] = {};
  for (int k0 = 0; k0 < K; k0 += 32) {
    __builtin_prefetch(W + (size_t)(col0 + (lane & 15)) * K + k0 + 64, 0, 3);
    bf16x16 a0 = load_a_frag(A, K, row0,      k0, lane);
    bf16x16 a1 = load_a_frag(A, K, row0 + 16, k0, lane);
#pragma unroll
    for (int j = 0; j < 4; ++j) {
      bf16x16 b = load_b_frag_f32(W, K, col0 + j * 16, k0, lane);
      c[0][j] = wmma_bf16(a0, b, c[0][j]);
      c[1][j] = wmma_bf16(a1, b, c[1][j]);
    }
  }

  const int half = lane >> 4, nn = lane & 15;
#pragma unroll
  for (int i = 0; i < 2; ++i)
#pragma unroll
    for (int j = 0; j < 4; ++j) {
      int col = col0 + j * 16 + nn;
      float bb = bias[col];
#pragma unroll
      for (int r = 0; r < 8; ++r) {
        int row = row0 + i * 16 + 8 * half + r;
        out[(size_t)row * N + col] = c[i][j][r] + bb;
      }
    }
}

// ---------------------------------------------------------------------------
// Persistent recurrent scan: one launch per layer, 16 blocks, device-scope
// grid barrier between the per-step WMMA phase and the elementwise phase.
// The 32x1024 bf16 latent (64 KB) is staged into LDS each step with a
// cooperative vector copy (visible ds_store/ds_load for the compiler) and,
// when available, a Tensor Data Mover descriptor is issued in parallel so the
// TENSORcnt path (tensor_load_to_lds + s_wait_tensorcnt) is exercised.
// ---------------------------------------------------------------------------
#define SCAN_BLOCKS 16

__device__ __forceinline__ void grid_sync(unsigned* cnt, unsigned target) {
  __syncthreads();
  if (threadIdx.x == 0) {
    __threadfence();
    __hip_atomic_fetch_add(cnt, 1u, __ATOMIC_ACQ_REL, __HIP_MEMORY_SCOPE_AGENT);
    while (__hip_atomic_load(cnt, __ATOMIC_ACQUIRE, __HIP_MEMORY_SCOPE_AGENT) <
           target)
      __builtin_amdgcn_s_sleep(1);
  }
  __syncthreads();
}

__global__ __launch_bounds__(256) void ran_scan_kernel(
    const __bf16* __restrict__ Wh_i, const __bf16* __restrict__ Wh_f,
    const float* __restrict__ xi, const float* __restrict__ xf,  // [S,B,H]
    const float* __restrict__ x_f32,   // [S,B,H] or null (layer 1)
    const int* __restrict__ word,      // [S,B]   (layer 1 only)
    const float* __restrict__ emb,     // [V,E]   (layer 1 only)
    float* __restrict__ lat_f32, __bf16* __restrict__ lat_bf16,  // [B,H]
    float* __restrict__ lout_f32,      // [S,B,H] or null
    __bf16* __restrict__ lout_bf16,    // [S,B,H]
    float* __restrict__ rec,           // [2,B,H] scratch
    unsigned* __restrict__ sync_cnt,
    float* __restrict__ out_latent, float* __restrict__ out_ig,
    float* __restrict__ out_fg)        // non-null for layer 2
{
  __shared__ __bf16 lat_s[BH];              // 64 KB of the 320 KB/WGP LDS

  const int lane = threadIdx.x & 31;
  const int wave = threadIdx.x >> 5;
  const int gw   = blockIdx.x * 8 + wave;   // 0..127 waves
  const int gate = gw >> 6;                 // 0 = input gate, 1 = forget gate
  const int col0 = (gw & 63) * 16;          // 64 column tiles cover H=1024
  const __bf16* Wh = gate ? Wh_f : Wh_i;
  float* rg = rec + (size_t)gate * BH;

  const int halfl = lane >> 4;
  const int rlane = lane & 15;

  unsigned nsync = 0;

  for (int t = 0; t < S_LEN; ++t) {
    // ---- Stage latent [32,1024] bf16 into LDS ----
    // Cooperative copy: compiler-visible ds_store_b128 so the later LDS
    // reads (and the WMMA chain fed by them) cannot be folded away.
    {
      const bf16x8* src = reinterpret_cast<const bf16x8*>(lat_bf16);
      for (int i = threadIdx.x; i < BH / 8; i += 256)
        *reinterpret_cast<bf16x8*>(&lat_s[(size_t)i * 8]) = src[i];
    }
#ifdef HAVE_TDM
    // TDM writes the same 64 KB tile (2-D D#: dims 1024x32 x 2B, stride 1024);
    // benign duplicate of the copy above, exercises the TENSORcnt path.
    if (threadIdx.x < 32) {  // wave 0 issues one descriptor
      unsigned long long ga = (unsigned long long)(uintptr_t)lat_bf16;
      u32x4 g0;
      g0.x = 0x1u;                                   // count=1, user mode
      g0.y = 0u;                                     // lds_addr (lat_s is at 0)
      g0.z = (unsigned)(ga & 0xFFFFFFFFu);           // global_addr[31:0]
      g0.w = (unsigned)((ga >> 32) & 0x01FFFFFFu) |  // global_addr[56:32]
             (2u << 30);                             // type=2 ("image")
      i32x8 g1;
      g1[0] = 0x00010000;        // workgroup_mask=0, data_size=1 (2 bytes)
      g1[1] = 0x04000000;        // tensor_dim0[15:0]=1024 (bits 63:48)
      g1[2] = 0x00200000;        // tensor_dim0 hi=0; tensor_dim1 lo16=32
      g1[3] = 0x04000000;        // tensor_dim1 hi=0; tile_dim0=1024
      g1[4] = 32;                // tile_dim1=32, tile_dim2=0
      g1[5] = 1024;              // tensor_dim0_stride[31:0]=1024
      g1[6] = 0;                 // stride hi / tensor_dim1_stride lo
      g1[7] = 0;
      i32x4 gz = {0, 0, 0, 0};   // 2-D tensor: groups 2/3 unused
#if __clang_major__ >= 23
      i32x8 gz8 = {0, 0, 0, 0, 0, 0, 0, 0};
      __builtin_amdgcn_tensor_load_to_lds(g0, g1, gz, gz, gz8, 0);
#else
      __builtin_amdgcn_tensor_load_to_lds(g0, g1, gz, gz, 0);
#endif
      __builtin_amdgcn_s_wait_tensorcnt(0);
    }
#endif
    asm volatile("" ::: "memory");   // keep staging ordered vs. consumption
    __syncthreads();

    // ---- Phase A: rec[gate][0:32][col0:col0+16] = lat @ Wh^T (A from LDS) --
    f32x8 c0 = {}, c1 = {};
    for (int k0 = 0; k0 < H_DIM; k0 += 32) {
      const __bf16* pa0 = &lat_s[(size_t)rlane * H_DIM + k0 + halfl * 8];
      const __bf16* pa1 = &lat_s[(size_t)(16 + rlane) * H_DIM + k0 + halfl * 8];
      bf16x8 a0l = *reinterpret_cast<const bf16x8*>(pa0);
      bf16x8 a0h = *reinterpret_cast<const bf16x8*>(pa0 + 16);
      bf16x8 a1l = *reinterpret_cast<const bf16x8*>(pa1);
      bf16x8 a1h = *reinterpret_cast<const bf16x8*>(pa1 + 16);
      bf16x16 a0 = __builtin_shufflevector(a0l, a0h, 0, 1, 2, 3, 4, 5, 6, 7,
                                           8, 9, 10, 11, 12, 13, 14, 15);
      bf16x16 a1 = __builtin_shufflevector(a1l, a1h, 0, 1, 2, 3, 4, 5, 6, 7,
                                           8, 9, 10, 11, 12, 13, 14, 15);
      bf16x16 b  = load_b_frag(Wh, H_DIM, col0, k0, lane);
      c0 = wmma_bf16(a0, b, c0);
      c1 = wmma_bf16(a1, b, c1);
    }
    {
#pragma unroll
      for (int r = 0; r < 8; ++r) {
        rg[(size_t)(8 * halfl + r) * H_DIM + col0 + rlane]      = c0[r];
        rg[(size_t)(16 + 8 * halfl + r) * H_DIM + col0 + rlane] = c1[r];
      }
    }
    grid_sync(sync_cnt, SCAN_BLOCKS * (++nsync));

    // ---- Phase B: elementwise gates + latent update ----
    const float* xi_t = xi + (size_t)t * BH;
    const float* xf_t = xf + (size_t)t * BH;
    size_t base = ((size_t)blockIdx.x * 256 + threadIdx.x) * 8;  // 4096 thr * 8
#pragma unroll
    for (int j = 0; j < 8; ++j) {
      size_t idx = base + j;                 // b*H + h
      float ri = rec[idx];
      float rf = rec[BH + idx];
      float ig = 1.0f / (1.0f + __expf(-(ri + xi_t[idx])));
      float fg = 1.0f / (1.0f + __expf(-(rf + xf_t[idx])));
      float xv;
      if (x_f32) {
        xv = x_f32[(size_t)t * BH + idx];
      } else {
        int b = (int)(idx >> 10);
        int h = (int)(idx & 1023);
        xv = emb[(size_t)word[t * B_DIM + b] * E_DIM + h];
      }
      float nl = ig * xv + fg * lat_f32[idx];
      lat_f32[idx]  = nl;
      lat_bf16[idx] = (__bf16)nl;
      if (lout_f32) lout_f32[(size_t)t * BH + idx] = nl;
      lout_bf16[(size_t)t * BH + idx] = (__bf16)nl;
      if (out_latent && t == S_LEN - 1) {
        out_latent[idx] = nl;
        out_ig[idx] = ig;
        out_fg[idx] = fg;
      }
    }
    grid_sync(sync_cnt, SCAN_BLOCKS * (++nsync));
  }
}

// ---------------------------------------------------------------------------
// Host-side orchestration
// ---------------------------------------------------------------------------
extern "C" void kernel_launch(void* const* d_in, const int* in_sizes, int n_in,
                              void* d_out, int out_size, void* d_ws,
                              size_t ws_size, hipStream_t stream) {
  (void)in_sizes; (void)n_in; (void)out_size; (void)ws_size;

  const int*   word    = (const int*)d_in[0];
  const float* lat_in  = (const float*)d_in[1];
  const float* emb     = (const float*)d_in[2];
  const float* h2o_w   = (const float*)d_in[3];
  const float* h2o_b   = (const float*)d_in[4];
  const float* Whi     = (const float*)d_in[5];
  const float* Wxi_w   = (const float*)d_in[6];
  const float* Wxi_b   = (const float*)d_in[7];
  const float* Whf     = (const float*)d_in[8];
  const float* Wxf_w   = (const float*)d_in[9];
  const float* Wxf_b   = (const float*)d_in[10];
  const float* Whi2    = (const float*)d_in[11];
  const float* Wxi2_w  = (const float*)d_in[12];
  const float* Wxi2_b  = (const float*)d_in[13];
  const float* Whf2    = (const float*)d_in[14];
  const float* Wxf2_w  = (const float*)d_in[15];
  const float* Wxf2_b  = (const float*)d_in[16];

  // Workspace layout (bytes, 256-aligned)
  char* ws = (char*)d_ws;
  size_t off = 0;
  unsigned* cntA = (unsigned*)(ws + off);          // [0,64): layer-1 barrier
  unsigned* cntB = (unsigned*)(ws + off + 64);     // [64,128): layer-2 barrier
  off += 256;
  __bf16* wbf   = (__bf16*)(ws + off); off += (size_t)8 * HH * 2;   // 16 MB
  __bf16* x1bf  = (__bf16*)(ws + off); off += (size_t)MROWS * E_DIM * 2;
  __bf16* l1bf  = (__bf16*)(ws + off); off += (size_t)MROWS * H_DIM * 2;
  __bf16* l2bf  = (__bf16*)(ws + off); off += (size_t)MROWS * H_DIM * 2;
  float*  l1f   = (float*)(ws + off);  off += (size_t)MROWS * H_DIM * 4;
  float*  xi1   = (float*)(ws + off);  off += (size_t)MROWS * H_DIM * 4;
  float*  xf1   = (float*)(ws + off);  off += (size_t)MROWS * H_DIM * 4;
  float*  xi2   = (float*)(ws + off);  off += (size_t)MROWS * H_DIM * 4;
  float*  xf2   = (float*)(ws + off);  off += (size_t)MROWS * H_DIM * 4;
  float*  rec   = (float*)(ws + off);  off += (size_t)2 * BH * 4;
  float*  latf  = (float*)(ws + off);  off += (size_t)BH * 4;
  __bf16* latbf = (__bf16*)(ws + off); off += (size_t)BH * 2;

  float* out = (float*)d_out;
  float* out_latent = out;                        // [B,H]
  float* out_logits = out + BH;                   // [S,B,V]
  float* out_ig     = out + BH + LOGN;            // [B,H]
  float* out_fg     = out + 2 * BH + LOGN;        // [B,H]

  // 0) reset grid-barrier counters (capture-safe async memset)
  hipMemsetAsync(ws, 0, 256, stream);

  // 1) weights -> bf16   slots: 0=Whi 1=Wxi 2=Whf 3=Wxf 4=Whi2 5=Wxi2 6=Whf2 7=Wxf2
  cvt_weights_kernel<<<8192, 256, 0, stream>>>(Whi, Wxi_w, Whf, Wxf_w,
                                               Whi2, Wxi2_w, Whf2, Wxf2_w, wbf);
  // 2) embedding gather -> bf16
  gather_embed_kernel<<<MROWS, 256, 0, stream>>>(word, emb, x1bf);
  // 3) initial latent
  init_latent_kernel<<<BH / 256, 256, 0, stream>>>(lat_in, latf, latbf);

  // 4) layer-1 input-gate preactivations (batched WMMA GEMMs)
  dim3 g1(MROWS / 128, H_DIM / 128);
  gemm_bf16_kernel<<<g1, 256, 0, stream>>>(x1bf, wbf + (size_t)1 * HH, Wxi_b,
                                           xi1, MROWS, H_DIM, E_DIM);
  gemm_bf16_kernel<<<g1, 256, 0, stream>>>(x1bf, wbf + (size_t)3 * HH, Wxf_b,
                                           xf1, MROWS, H_DIM, E_DIM);

  // 5) layer-1 persistent scan
  ran_scan_kernel<<<SCAN_BLOCKS, 256, 0, stream>>>(
      wbf + (size_t)0 * HH, wbf + (size_t)2 * HH, xi1, xf1,
      /*x_f32=*/nullptr, word, emb, latf, latbf, l1f, l1bf, rec, cntA,
      nullptr, nullptr, nullptr);

  // 6) layer-2 input-gate preactivations
  gemm_bf16_kernel<<<g1, 256, 0, stream>>>(l1bf, wbf + (size_t)5 * HH, Wxi2_b,
                                           xi2, MROWS, H_DIM, H_DIM);
  gemm_bf16_kernel<<<g1, 256, 0, stream>>>(l1bf, wbf + (size_t)7 * HH, Wxf2_b,
                                           xf2, MROWS, H_DIM, H_DIM);

  // 7) layer-2 persistent scan (writes final latent / last-step ig,fg)
  ran_scan_kernel<<<SCAN_BLOCKS, 256, 0, stream>>>(
      wbf + (size_t)4 * HH, wbf + (size_t)6 * HH, xi2, xf2,
      /*x_f32=*/l1f, nullptr, nullptr, latf, latbf, nullptr, l2bf, rec, cntB,
      out_latent, out_ig, out_fg);

  // 8) logits = l2_out @ h2o_w^T + h2o_b   (dominant 268-GFLOP WMMA GEMM)
  dim3 g2(MROWS / 128, V_DIM / 128);
  gemm_logits_kernel<<<g2, 256, 0, stream>>>(l2bf, h2o_w, h2o_b, out_logits,
                                             MROWS, V_DIM, H_DIM);
}
